// EnergyFunction_42949673431
// MI455X (gfx1250) — matrix operand order repaired and checked by reading, not compiled, not run on hardware
//
#include <hip/hip_runtime.h>

typedef __attribute__((ext_vector_type(16))) _Float16 v16h;
typedef __attribute__((ext_vector_type(8)))  float    v8f;

#define BATCH   2
#define SEQ     2048
#define DMODEL  1024
#define NH      8
#define HD      128
#define MTOT    (BATCH*SEQ)     // 4096

// ---------------------------------------------------------------------------
// Fragment loaders (layouts per CDNA5 ISA §7.12.2, wave32)
//  A 16x32 f16 : lane(0-15)=row m, runs K = {8*hh .. 8*hh+7} and {16+8*hh ..}
//  B 32x16 f16 : lane holds column n=lane&15, runs K = {8*hh .. 8*hh+7} and {16+8*hh ..}
// ---------------------------------------------------------------------------
__device__ __forceinline__ v16h frag_a(const _Float16* p) {
  union { v16h v; uint4 u[2]; } t;
  t.u[0] = *(const uint4*)(p);        // K run 0..7
  t.u[1] = *(const uint4*)(p + 16);   // K run 16..23
  return t.v;
}
__device__ __forceinline__ v16h frag_b(const _Float16* p) {
  union { v16h v; uint4 u[2]; } t;
  t.u[0] = *(const uint4*)(p);        // K run 0..7
  t.u[1] = *(const uint4*)(p + 16);   // K run 16..23
  return t.v;
}
__device__ __forceinline__ v8f wmma_f16(v16h a, v16h b, v8f c) {
  return __builtin_amdgcn_wmma_f32_16x16x32_f16(false, a, false, b,
                                                (short)0, c, false, false);
}

// ---------------------------------------------------------------------------
// 1) f32 -> f16 cast
// ---------------------------------------------------------------------------
__global__ void cast_f16_kernel(const float* __restrict__ in,
                                _Float16* __restrict__ out, int n) {
  int i = blockIdx.x * blockDim.x + threadIdx.x;
  if (i < n) out[i] = (_Float16)in[i];
}

// ---------------------------------------------------------------------------
// 2) C[M,N] = A[M,K] @ W[N,K]^T (torch Linear), optional f16 C-add, f16/f32 out
//    Block: 256 thr = 8 waves in 2(M) x 4(N); block tile 64(M) x 256(N)
//    Wave tile 32(M) x 64(N): 2 A-frags + 4 B-frags -> 8 WMMAs per k-step
// ---------------------------------------------------------------------------
template <bool ADD_C, bool OUT_F32>
__global__ void __launch_bounds__(256)
gemm_xwt_kernel(const _Float16* __restrict__ A,
                const _Float16* __restrict__ W,
                const _Float16* __restrict__ Cadd,
                _Float16* __restrict__ out16,
                float* __restrict__ out32,
                int M, int N, int K) {
  const int lane  = threadIdx.x & 31;
  const int wave  = threadIdx.x >> 5;
  const int hh    = lane >> 4;       // lane half
  const int ln    = lane & 15;
  const int row0  = blockIdx.y * 64  + (wave & 1) * 32;
  const int col0  = blockIdx.x * 256 + (wave >> 1) * 64;

  v8f acc[2][4] = {};
  const _Float16* Arow0 = A + (size_t)(row0 + ln) * K;        // rows row0..+15
  const _Float16* Arow1 = A + (size_t)(row0 + 16 + ln) * K;   // rows +16..+31

  for (int k0 = 0; k0 < K; k0 += 32) {
    if (k0 + 256 < K) {
      __builtin_prefetch(Arow0 + k0 + 256, 0, 0);
      __builtin_prefetch(Arow1 + k0 + 256, 0, 0);
    }
    v16h a0 = frag_a(Arow0 + k0 + 8 * hh);
    v16h a1 = frag_a(Arow1 + k0 + 8 * hh);
#pragma unroll
    for (int nf = 0; nf < 4; ++nf) {
      const _Float16* Wrow = W + (size_t)(col0 + nf * 16 + ln) * K;
      v16h b = frag_b(Wrow + k0 + 8 * hh);
      acc[0][nf] = wmma_f16(a0, b, acc[0][nf]);
      acc[1][nf] = wmma_f16(a1, b, acc[1][nf]);
    }
  }
#pragma unroll
  for (int mt = 0; mt < 2; ++mt) {
#pragma unroll
    for (int nf = 0; nf < 4; ++nf) {
      const int col = col0 + nf * 16 + ln;
#pragma unroll
      for (int r = 0; r < 8; ++r) {
        const int row = row0 + mt * 16 + r + 8 * hh;  // C layout: m = r + 8*hh
        size_t idx = (size_t)row * N + col;
        float v = acc[mt][nf][r];
        if constexpr (ADD_C) v += (float)Cadd[idx];
        if constexpr (OUT_F32) out32[idx] = v;
        else                   out16[idx] = (_Float16)v;
      }
    }
  }
}

// ---------------------------------------------------------------------------
// 3) V [B,S,H,hd] (as [B,S,D]) -> VT [B,H,hd,S]
// ---------------------------------------------------------------------------
__global__ void transpose_v_kernel(const _Float16* __restrict__ vals,
                                   _Float16* __restrict__ vt, int total) {
  int t = blockIdx.x * blockDim.x + threadIdx.x;
  if (t >= total) return;
  int s    = t % SEQ;
  int rest = t / SEQ;
  int d    = rest % HD;
  int bh   = rest / HD;
  int b = bh / NH, h = bh % NH;
  vt[t] = vals[(size_t)(b * SEQ + s) * DMODEL + h * HD + d];
}

// ---------------------------------------------------------------------------
// 4) Flash-style causal attention:
//    weights = softmax(-(K K^T)/sqrt(hd)) (causal), pairwise = weights @ V
//    Block: 128 thr = 4 waves; wave handles one 16-query tile.
//    grid = (S/64, B*H)
// ---------------------------------------------------------------------------
__global__ void __launch_bounds__(128)
flash_attn_kernel(const _Float16* __restrict__ Kmat,   // [B,S,D] keys
                  const _Float16* __restrict__ VT,     // [B,H,HD,S]
                  _Float16* __restrict__ out) {        // [B,S,D] pairwise
  __shared__ __align__(16) _Float16 plds[4][16][32];   // per-wave P tile

  const int lane = threadIdx.x & 31;
  const int wave = threadIdx.x >> 5;
  const int hh   = lane >> 4;
  const int ln   = lane & 15;
  const int b = blockIdx.y / NH, h = blockIdx.y % NH;
  const int qTile = blockIdx.x * 64 + wave * 16;

  const _Float16* Kp  = Kmat + (size_t)b * SEQ * DMODEL + h * HD;
  const _Float16* VTp = VT + (size_t)blockIdx.y * HD * SEQ;

  const float scale = -0.08838834764831845f;   // -1/sqrt(128)
  float mrun[8], lrun[8];
  v8f O[8] = {};
#pragma unroll
  for (int r = 0; r < 8; ++r) { mrun[r] = -3.0e38f; lrun[r] = 0.0f; }

  const _Float16* Aq = Kp + (size_t)(qTile + ln) * DMODEL;  // query rows
  const int kkEnd = qTile + 16;                              // causal extent

  for (int kk = 0; kk < kkEnd; kk += 32) {
    // ---- scores for 32 keys: two 16x16 C tiles --------------------------
    v8f s0 = {}, s1 = {};
#pragma unroll
    for (int d0 = 0; d0 < HD; d0 += 32) {
      v16h a  = frag_a(Aq + d0 + 8 * hh);
      v16h b0 = frag_b(Kp + (size_t)(kk + ln) * DMODEL      + d0 + 8 * hh);
      v16h b1 = frag_b(Kp + (size_t)(kk + 16 + ln) * DMODEL + d0 + 8 * hh);
      s0 = wmma_f16(a, b0, s0);
      s1 = wmma_f16(a, b1, s1);
    }
    // ---- causal mask + online softmax (rows live across 16-lane halves) --
    float p0[8], p1[8], alpha[8];
#pragma unroll
    for (int r = 0; r < 8; ++r) {
      const int q = qTile + r + 8 * hh;
      float e0 = (kk + ln      <= q) ? s0[r] * scale : -3.0e38f;
      float e1 = (kk + 16 + ln <= q) ? s1[r] * scale : -3.0e38f;
      float mx = fmaxf(e0, e1);
#pragma unroll
      for (int off = 1; off < 16; off <<= 1)
        mx = fmaxf(mx, __shfl_xor(mx, off, 32));
      const float mnew = fmaxf(mrun[r], mx);
      alpha[r] = __expf(mrun[r] - mnew);
      const float q0 = __expf(e0 - mnew);
      const float q1 = __expf(e1 - mnew);
      float rs = q0 + q1;
#pragma unroll
      for (int off = 1; off < 16; off <<= 1)
        rs += __shfl_xor(rs, off, 32);
      lrun[r] = lrun[r] * alpha[r] + rs;
      mrun[r] = mnew;
      p0[r] = q0; p1[r] = q1;
    }
    // ---- rescale O by alpha ---------------------------------------------
#pragma unroll
    for (int j = 0; j < 8; ++j)
#pragma unroll
      for (int r = 0; r < 8; ++r) O[j][r] *= alpha[r];
    // ---- C-layout probs -> A-fragment via per-wave LDS slab --------------
#pragma unroll
    for (int r = 0; r < 8; ++r) {
      plds[wave][r + 8 * hh][ln]      = (_Float16)p0[r];
      plds[wave][r + 8 * hh][16 + ln] = (_Float16)p1[r];
    }
    __builtin_amdgcn_wave_barrier();
    asm volatile("s_wait_dscnt 0x0" ::: "memory");  // LDS in-order per wave
    union { v16h v; uint4 u[2]; } ap;
    ap.u[0] = *(const uint4*)&plds[wave][ln][8 * hh];
    ap.u[1] = *(const uint4*)&plds[wave][ln][16 + 8 * hh];
    __builtin_amdgcn_wave_barrier();
    // ---- O += P @ V (hd in 8 N-tiles of 16) ------------------------------
#pragma unroll
    for (int j = 0; j < 8; ++j) {
      v16h bv = frag_b(VTp + (size_t)(j * 16 + ln) * SEQ + kk + 8 * hh);
      O[j] = wmma_f16(ap.v, bv, O[j]);
    }
  }
  // ---- normalize and store pairwise [B,S,D] ------------------------------
#pragma unroll
  for (int r = 0; r < 8; ++r) {
    const float inv = 1.0f / lrun[r];
    const int q = qTile + r + 8 * hh;
    _Float16* orow = out + (size_t)(b * SEQ + q) * DMODEL + h * HD;
#pragma unroll
    for (int j = 0; j < 8; ++j)
      orow[j * 16 + ln] = (_Float16)(O[j][r] * inv);
  }
}

// ---------------------------------------------------------------------------
// Launcher
// ---------------------------------------------------------------------------
extern "C" void kernel_launch(void* const* d_in, const int* in_sizes, int n_in,
                              void* d_out, int out_size, void* d_ws, size_t ws_size,
                              hipStream_t stream) {
  const float* x     = (const float*)d_in[0];
  const float* Wk    = (const float*)d_in[1];
  const float* Wv    = (const float*)d_in[2];
  const float* Wself = (const float*)d_in[3];
  const float* Wout  = (const float*)d_in[4];
  float* out = (float*)d_out;

  const size_t nX = (size_t)MTOT * DMODEL;       // 4,194,304
  const size_t nW = (size_t)DMODEL * DMODEL;     // 1,048,576

  _Float16* x16   = (_Float16*)d_ws;
  _Float16* wk16  = x16   + nX;
  _Float16* wv16  = wk16  + nW;
  _Float16* ws16  = wv16  + nW;
  _Float16* wo16  = ws16  + nW;
  _Float16* keys  = wo16  + nW;   // [B,S,D]; later reused as mid16
  _Float16* vals  = keys  + nX;   // [B,S,D]; later reused as attn16
  _Float16* vt    = vals  + nX;   // [B,H,HD,S]
  _Float16* attn  = vals;         // alias: vals dead after transpose
  _Float16* mid   = keys;         // alias: keys dead after attention

  // 1) casts
  cast_f16_kernel<<<(int)((nX + 255) / 256), 256, 0, stream>>>(x, x16, (int)nX);
  cast_f16_kernel<<<(int)((nW + 255) / 256), 256, 0, stream>>>(Wk,    wk16, (int)nW);
  cast_f16_kernel<<<(int)((nW + 255) / 256), 256, 0, stream>>>(Wv,    wv16, (int)nW);
  cast_f16_kernel<<<(int)((nW + 255) / 256), 256, 0, stream>>>(Wself, ws16, (int)nW);
  cast_f16_kernel<<<(int)((nW + 255) / 256), 256, 0, stream>>>(Wout,  wo16, (int)nW);

  // 2) keys = x @ Wk^T ; vals = x @ Wv^T
  dim3 gGemm(DMODEL / 256, MTOT / 64);
  gemm_xwt_kernel<false, false><<<gGemm, 256, 0, stream>>>(
      x16, wk16, nullptr, keys, nullptr, MTOT, DMODEL, DMODEL);
  gemm_xwt_kernel<false, false><<<gGemm, 256, 0, stream>>>(
      x16, wv16, nullptr, vals, nullptr, MTOT, DMODEL, DMODEL);

  // 3) VT[b,h,d,s]
  transpose_v_kernel<<<(int)((nX + 255) / 256), 256, 0, stream>>>(vals, vt, (int)nX);

  // 4) pairwise = causal_softmax(-K K^T / sqrt(hd)) @ V
  dim3 gAttn(SEQ / 64, BATCH * NH);
  flash_attn_kernel<<<gAttn, 128, 0, stream>>>(keys, vt, attn);

  // 5) mid = pairwise + x @ Wself^T
  gemm_xwt_kernel<true, false><<<gGemm, 256, 0, stream>>>(
      x16, ws16, attn, mid, nullptr, MTOT, DMODEL, DMODEL);

  // 6) out = mid @ Wout^T  (f32)
  gemm_xwt_kernel<false, true><<<gGemm, 256, 0, stream>>>(
      mid, wo16, nullptr, nullptr, out, MTOT, DMODEL, DMODEL);
}
